// CNNFocusing_13606456394167
// MI455X (gfx1250) — matrix operand order repaired and checked by reading, not compiled, not run on hardware
//
#include <hip/hip_runtime.h>
#include <hip/hip_bf16.h>

typedef __attribute__((ext_vector_type(16))) _Float16 v16h;
typedef __attribute__((ext_vector_type(8)))  float    v8f;
typedef __attribute__((ext_vector_type(4)))  float    f32x4;
typedef __attribute__((ext_vector_type(4)))  _Float16 h16x4;
typedef __attribute__((ext_vector_type(8)))  _Float16 h16x8;

#define NGROUP 72

// ---------------------------------------------------------------------------
// Grouped conv + bias + ReLU + 2x2 maxpool, implicit-GEMM on WMMA.
// One workgroup per (batch, group). 256 threads = 8 waves (wave32).
//   M = 16 conv output positions (tile), N = out-channels (padded to 16),
//   K = IC*KS*KS padded to multiple of 32.
// InF32: input is f32 (first conv) vs f16 (later stages).
// All staging is vectorized (b128 global loads, packed cvt, b64/b128 LDS
// stores); extents are compile-time divisible by the vector widths.
// ---------------------------------------------------------------------------
template <int IC, int OC, int KS, int IH, int InF32>
__global__ __launch_bounds__(256) void conv_relu_pool_wmma(
    const void* __restrict__ in_v, const float* __restrict__ w,
    const float* __restrict__ bias, _Float16* __restrict__ out)
{
    constexpr int IW     = IH;
    constexpr int OHW    = IH - KS + 1;       // conv output edge
    constexpr int NPOS   = OHW * OHW;         // conv output positions
    constexpr int PH     = OHW / 2;           // pooled edge (floor, VALID)
    constexpr int KDIM   = IC * KS * KS;      // GEMM K
    constexpr int KSTEPS = (KDIM + 31) / 32;  // padded K steps
    constexpr int NT     = (OC + 15) / 16;    // N tiles
    constexpr int NTILES = (NPOS + 15) / 16;  // M tiles
    constexpr int INELEM = IC * IH * IW;
    constexpr int WELEM  = OC * KDIM;
    static_assert(INELEM % 4 == 0 && WELEM % 4 == 0, "vector staging");

    __shared__ _Float16 s_in[INELEM];
    __shared__ _Float16 s_w[WELEM];
    __shared__ float    s_bias[OC];
    __shared__ _Float16 s_conv[OC * NPOS];

    const int b   = blockIdx.x / NGROUP;
    const int g   = blockIdx.x % NGROUP;
    const int tid = threadIdx.x;

    // ---- Stage input patch for this (batch, group) into LDS as f16 ----
    if constexpr (InF32) {
        const f32x4* src = (const f32x4*)((const float*)in_v +
                                          (size_t)(b * NGROUP + g) * INELEM);
        h16x4* dst = (h16x4*)s_in;
        for (int i = tid; i < INELEM / 4; i += 256)
            dst[i] = __builtin_convertvector(src[i], h16x4);
    } else {
        const h16x8* src = (const h16x8*)((const _Float16*)in_v +
                                          (size_t)(b * NGROUP + g) * INELEM);
        h16x8* dst = (h16x8*)s_in;
        for (int i = tid; i < INELEM / 8; i += 256)
            dst[i] = src[i];
    }
    // ---- Stage group weights (f32 -> f16) and biases ----
    {
        const f32x4* src = (const f32x4*)(w + (size_t)g * WELEM);
        h16x4* dst = (h16x4*)s_w;
        for (int i = tid; i < WELEM / 4; i += 256)
            dst[i] = __builtin_convertvector(src[i], h16x4);
        if (tid < OC) s_bias[tid] = bias[g * OC + tid];
    }
    __syncthreads();

    const int lane   = tid & 31;
    const int wave   = tid >> 5;
    const int mrow   = lane & 15;            // M (A) or N (B/D) within tile
    const int khalfA = (lane >= 16) ? 16 : 0;// A-fragment K half offset
    const int khalfB = (lane >= 16) ? 16 : 0;// B-fragment K half offset
    const int drow   = (lane >= 16) ? 8 : 0; // D-fragment row offset

    for (int tile = wave; tile < NTILES; tile += 8) {
        v8f acc[NT];
        const v8f vzero = {0.f, 0.f, 0.f, 0.f, 0.f, 0.f, 0.f, 0.f};
        #pragma unroll
        for (int nt = 0; nt < NT; ++nt) acc[nt] = vzero;

        const int pos   = tile * 16 + mrow;
        const bool mval = (pos < NPOS);
        const int oy = mval ? pos / OHW : 0;
        const int ox = mval ? pos % OHW : 0;
        const int abase = oy * IW + ox;      // all K offsets are immediates

        #pragma unroll
        for (int ks = 0; ks < KSTEPS; ++ks) {
            // A fragment: im2col gather, immediate offsets from abase.
            v16h a;
            #pragma unroll
            for (int i = 0; i < 16; ++i) {
                const int K = ks * 32 + khalfA + i;
                _Float16 v = (_Float16)0.f;
                if (K < KDIM && mval) {
                    const int ic = K / (KS * KS);
                    const int r  = K % (KS * KS);
                    v = s_in[abase + ic * IH * IW + (r / KS) * IW + (r % KS)];
                }
                a[i] = v;
            }
            // B fragments per N-tile (contiguous K -> b128 DS loads).
            #pragma unroll
            for (int nt = 0; nt < NT; ++nt) {
                v16h bm;
                const int oc = nt * 16 + mrow;
                #pragma unroll
                for (int i = 0; i < 16; ++i) {
                    const int K = ks * 32 + khalfB + i;
                    bm[i] = (K < KDIM && oc < OC) ? s_w[oc * KDIM + K]
                                                  : (_Float16)0.f;
                }
                acc[nt] = __builtin_amdgcn_wmma_f32_16x16x32_f16(
                    false, a, false, bm, (short)0, acc[nt], false, false);
            }
        }
        // Bias + ReLU, scatter D fragment into LDS conv buffer.
        #pragma unroll
        for (int nt = 0; nt < NT; ++nt) {
            const int oc = nt * 16 + mrow;
            #pragma unroll
            for (int j = 0; j < 8; ++j) {
                const int row = tile * 16 + j + drow;
                if (row < NPOS && oc < OC) {
                    float val = acc[nt][j] + s_bias[oc];
                    val = val > 0.f ? val : 0.f;
                    s_conv[oc * NPOS + row] = (_Float16)val;
                }
            }
        }
    }
    __syncthreads();

    // ---- 2x2 maxpool (VALID) -> global f16, channel-major layout ----
    for (int i = tid; i < OC * PH * PH; i += 256) {
        const int oc = i / (PH * PH);
        const int r  = i % (PH * PH);
        const int py = r / PH, px = r % PH;
        const _Float16* p = &s_conv[oc * NPOS + (2 * py) * OHW + 2 * px];
        const float m0 = (float)p[0], m1 = (float)p[1];
        const float m2 = (float)p[OHW], m3 = (float)p[OHW + 1];
        const float m  = fmaxf(fmaxf(m0, m1), fmaxf(m2, m3));
        out[(size_t)((b * NGROUP + g) * OC + oc) * (PH * PH) + r] = (_Float16)m;
    }
}

// ---------------------------------------------------------------------------
// Block-diagonal linear1: per group, (64 batch x 288) @ (288 x 64) -> 64x64.
// One workgroup per group; 8 waves cover 4x4 WMMA tiles, K = 9 steps of 32.
// Two N-halves so static LDS stays under 64 KB.
// ---------------------------------------------------------------------------
__global__ __launch_bounds__(256) void lin1_wmma(
    const _Float16* __restrict__ h, const float* __restrict__ w,
    _Float16* __restrict__ out1)
{
    __shared__ _Float16 s_a[64 * 288];  // 36 KB activations
    __shared__ _Float16 s_b[32 * 288];  // 18 KB half weight block

    const int g   = blockIdx.x;
    const int tid = threadIdx.x;

    // A block: 64 rows x 288 cols, rows strided in global -> b128 per 8 halves
    for (int i = tid; i < 64 * (288 / 8); i += 256) {
        const int m = i / 36, kv = i % 36;
        ((h16x8*)s_a)[i] =
            *(const h16x8*)(h + (size_t)m * 20736 + g * 288 + kv * 8);
    }

    const int lane   = tid & 31;
    const int wave   = tid >> 5;
    const int mrow   = lane & 15;
    const int khalfA = (lane >= 16) ? 16 : 0;
    const int khalfB = (lane >= 16) ? 16 : 0;
    const int drow   = (lane >= 16) ? 8 : 0;

    for (int nh = 0; nh < 2; ++nh) {
        __syncthreads();
        // B half-block: 32 rows x 288 cols, f32 -> f16 packed conversion.
        for (int i = tid; i < 32 * (288 / 4); i += 256) {
            const int nl = i / 72, kv = i % 72;
            const int n  = nh * 32 + nl;
            f32x4 v = *(const f32x4*)(w + (size_t)(g * 64 + n) * 20736 +
                                      g * 288 + kv * 4);
            ((h16x4*)s_b)[nl * 72 + kv] = __builtin_convertvector(v, h16x4);
        }
        __syncthreads();

        // 8 tiles this half: mt in 0..3, nlocal in 0..1 -> one per wave.
        const int mt = wave >> 1, nlocal = wave & 1;
        v8f acc = {0.f, 0.f, 0.f, 0.f, 0.f, 0.f, 0.f, 0.f};
        #pragma unroll
        for (int ks = 0; ks < 9; ++ks) {
            v16h a, bm;
            #pragma unroll
            for (int i = 0; i < 16; ++i) {
                const int K = ks * 32 + khalfA + i;
                a[i] = s_a[(mt * 16 + mrow) * 288 + K];
            }
            #pragma unroll
            for (int i = 0; i < 16; ++i) {
                const int K = ks * 32 + khalfB + i;
                bm[i] = s_b[(nlocal * 16 + mrow) * 288 + K];
            }
            acc = __builtin_amdgcn_wmma_f32_16x16x32_f16(
                false, a, false, bm, (short)0, acc, false, false);
        }
        const int n = (nh * 2 + nlocal) * 16 + mrow;
        #pragma unroll
        for (int j = 0; j < 8; ++j) {
            const int bb = mt * 16 + j + drow;
            out1[bb * 4608 + g * 64 + n] = (_Float16)acc[j];
        }
    }
}

// ---------------------------------------------------------------------------
// Block-diagonal linear2 + bias: 64 batch x 144 outputs, 64-length dots.
// Tiny (0.6 MFLOP) -> plain VALU in f32, b128 reads.
// ---------------------------------------------------------------------------
__global__ __launch_bounds__(256) void lin2_kernel(
    const _Float16* __restrict__ out1, const float* __restrict__ w,
    const float* __restrict__ bias, float* __restrict__ out)
{
    const int gid = blockIdx.x * blockDim.x + threadIdx.x;
    if (gid >= 64 * 144) return;
    const int b = gid / 144, o = gid % 144;
    const int g = o >> 1;
    const h16x8* hp8 = (const h16x8*)(out1 + (size_t)b * 4608 + g * 64);
    const f32x4* wp4 = (const f32x4*)(w + (size_t)o * 4608 + g * 64);
    float s = bias[o];
    #pragma unroll
    for (int c8 = 0; c8 < 8; ++c8) {
        const h16x8 hv = hp8[c8];
        const f32x4 w0 = wp4[2 * c8 + 0];
        const f32x4 w1 = wp4[2 * c8 + 1];
        s += (float)hv[0] * w0[0] + (float)hv[1] * w0[1] +
             (float)hv[2] * w0[2] + (float)hv[3] * w0[3] +
             (float)hv[4] * w1[0] + (float)hv[5] * w1[1] +
             (float)hv[6] * w1[2] + (float)hv[7] * w1[3];
    }
    out[gid] = s;
}

extern "C" void kernel_launch(void* const* d_in, const int* in_sizes, int n_in,
                              void* d_out, int out_size, void* d_ws, size_t ws_size,
                              hipStream_t stream)
{
    (void)in_sizes; (void)n_in; (void)out_size; (void)ws_size;
    const float* x    = (const float*)d_in[0];
    const float* w1   = (const float*)d_in[1];
    const float* b1   = (const float*)d_in[2];
    const float* w2   = (const float*)d_in[3];
    const float* b2   = (const float*)d_in[4];
    const float* w3   = (const float*)d_in[5];
    const float* b3   = (const float*)d_in[6];
    const float* lw1  = (const float*)d_in[7];
    const float* lw2  = (const float*)d_in[8];
    const float* lb2  = (const float*)d_in[9];

    // Workspace carve-up (f16 intermediates), 256B aligned.
    auto alignup = [](size_t v) { return (v + 255) & ~(size_t)255; };
    char* ws = (char*)d_ws;
    size_t off = 0;
    _Float16* p1 = (_Float16*)(ws + off); off += alignup((size_t)64 * 576 * 21 * 21 * 2);
    _Float16* p2 = (_Float16*)(ws + off); off += alignup((size_t)64 * 1152 * 9 * 9 * 2);
    _Float16* hb = (_Float16*)(ws + off); off += alignup((size_t)64 * 2304 * 3 * 3 * 2);
    _Float16* o1 = (_Float16*)(ws + off); off += alignup((size_t)64 * 4608 * 2);

    const dim3 blk(256);
    const dim3 grid_conv(64 * NGROUP);

    // conv1: 3->8 ch/group, k=5, 46x46 in, f32 input
    conv_relu_pool_wmma<3, 8, 5, 46, 1><<<grid_conv, blk, 0, stream>>>(x, w1, b1, p1);
    // conv2: 8->16 ch/group, k=3, 21x21 in, f16 input
    conv_relu_pool_wmma<8, 16, 3, 21, 0><<<grid_conv, blk, 0, stream>>>(p1, w2, b2, p2);
    // conv3: 16->32 ch/group, k=3, 9x9 in, f16 input
    conv_relu_pool_wmma<16, 32, 3, 9, 0><<<grid_conv, blk, 0, stream>>>(p2, w3, b3, hb);
    // block-diagonal linear1 (per-group 64x64x288 GEMM on WMMA)
    lin1_wmma<<<72, blk, 0, stream>>>(hb, lw1, o1);
    // block-diagonal linear2 + bias
    lin2_kernel<<<(64 * 144 + 255) / 256, blk, 0, stream>>>(o1, lw2, lb2, (float*)d_out);
}
